// ConvBertEncoder_283467841605
// MI455X (gfx1250) — matrix operand order repaired and checked by reading, not compiled, not run on hardware
//
#include <hip/hip_runtime.h>
#include <math.h>

// ---------------------------------------------------------------------------
// ConvBERT encoder layer on MI455X (gfx1250).
//
// Roofline: ~116 GFLOP of GEMM vs ~600MB of HBM traffic (~25us @ 23.3TB/s)
// -> compute-bound -> every GEMM runs on v_wmma_f32_16x16x32_bf16 (fp32
// operands RNE-converted to bf16 while staging through LDS, f32 accumulate).
//
// GEMM: 128x128 block tile, 256 threads = 8 wave32, each wave owns 32x64
// (2x4 WMMA tiles, 8 WMMAs per K-step of 32). Double-buffered LDS stages:
// global float4 loads for step k+32 are issued before the WMMAs of step k.
// B is addressed with generic (row,col) strides so Q@K^T needs no transpose
// pass; both stride cases vectorize to global_load_b128.
//
// Attention runs per (b,h): each 2048x2048 score matrix is 16MB and stays
// resident in the 192MB L2 between score GEMM, masked softmax and ctx GEMM.
// ---------------------------------------------------------------------------

typedef __attribute__((ext_vector_type(16))) __bf16          v16bf;
typedef __attribute__((ext_vector_type(8)))  float           v8f;
typedef __attribute__((ext_vector_type(8)))  unsigned short  us8;
typedef __attribute__((ext_vector_type(4)))  unsigned short  us4;
typedef __attribute__((ext_vector_type(4)))  unsigned int    u32x4;
typedef __attribute__((ext_vector_type(8)))  int             i32x8;
typedef __attribute__((ext_vector_type(4)))  int             i32x4;

#define BM 128
#define BN 128
#define BK 32
#define LDW 40   // padded LDS row stride in ushorts (80B rows; 8B/16B aligned)

__device__ __forceinline__ unsigned short f2bf(float f) {
  unsigned int u = __float_as_uint(f);
  u += 0x7FFFu + ((u >> 16) & 1u);          // round-to-nearest-even
  return (unsigned short)(u >> 16);
}

// ---- GEMM staging/compute macros (textual so shared arrays keep addrspace(3))

#define GEMM_GLOAD(K0)                                                         \
  {                                                                            \
    _Pragma("unroll") for (int it = 0; it < 4; ++it) {                         \
      int idx = tid + 256 * it;                                                \
      int r = idx >> 3, c4 = idx & 7;                                          \
      areg[it] = *(const float4*)&A[(long long)(mb + r) * lda + (K0) + c4 * 4];\
    }                                                                          \
    if (bcs == 1) {                                                            \
      _Pragma("unroll") for (int it = 0; it < 4; ++it) {                       \
        int idx = tid + 256 * it;                                              \
        int kk = idx >> 5, n4 = idx & 31;                                      \
        float4 z = make_float4(0.f, 0.f, 0.f, 0.f);                            \
        if (nb + n4 * 4 < Ncols)                                               \
          z = *(const float4*)&Bm[((long long)(K0) + kk) * brs + (nb + n4 * 4)];\
        breg[it] = z;                                                          \
      }                                                                        \
    } else {                                                                   \
      _Pragma("unroll") for (int it = 0; it < 4; ++it) {                       \
        int idx = tid + 256 * it;                                              \
        int n = idx >> 3, k4 = idx & 7;                                        \
        float4 z = make_float4(0.f, 0.f, 0.f, 0.f);                            \
        if (nb + n < Ncols)                                                    \
          z = *(const float4*)&Bm[(long long)(nb + n) * bcs + (K0) + k4 * 4];  \
        breg[it] = z;                                                          \
      }                                                                        \
    }                                                                          \
  }

#define GEMM_LSTORE(BUF)                                                       \
  {                                                                            \
    _Pragma("unroll") for (int it = 0; it < 4; ++it) {                         \
      int idx = tid + 256 * it;                                                \
      int r = idx >> 3, c4 = idx & 7;                                          \
      us4 pa;                                                                  \
      pa[0] = f2bf(areg[it].x); pa[1] = f2bf(areg[it].y);                      \
      pa[2] = f2bf(areg[it].z); pa[3] = f2bf(areg[it].w);                      \
      *(us4*)&As[BUF][r * LDW + c4 * 4] = pa;                                  \
    }                                                                          \
    if (bcs == 1) {                                                            \
      _Pragma("unroll") for (int it = 0; it < 4; ++it) {                       \
        int idx = tid + 256 * it;                                              \
        int kk = idx >> 5, n4 = idx & 31;                                      \
        Bs[BUF][(n4 * 4 + 0) * LDW + kk] = f2bf(breg[it].x);                   \
        Bs[BUF][(n4 * 4 + 1) * LDW + kk] = f2bf(breg[it].y);                   \
        Bs[BUF][(n4 * 4 + 2) * LDW + kk] = f2bf(breg[it].z);                   \
        Bs[BUF][(n4 * 4 + 3) * LDW + kk] = f2bf(breg[it].w);                   \
      }                                                                        \
    } else {                                                                   \
      _Pragma("unroll") for (int it = 0; it < 4; ++it) {                       \
        int idx = tid + 256 * it;                                              \
        int n = idx >> 3, k4 = idx & 7;                                        \
        us4 pb;                                                                \
        pb[0] = f2bf(breg[it].x); pb[1] = f2bf(breg[it].y);                    \
        pb[2] = f2bf(breg[it].z); pb[3] = f2bf(breg[it].w);                    \
        *(us4*)&Bs[BUF][n * LDW + k4 * 4] = pb;                                \
      }                                                                        \
    }                                                                          \
  }

#define GEMM_COMPUTE(BUF)                                                      \
  {                                                                            \
    v16bf afr[2], bfr[4];                                                      \
    _Pragma("unroll") for (int i = 0; i < 2; ++i) {                            \
      union { us8 h[2]; v16bf v; } ua;                                         \
      int r = wm + i * 16 + mr;                                                \
      ua.h[0] = *(const us8*)&As[BUF][r * LDW + half * 8];                     \
      ua.h[1] = *(const us8*)&As[BUF][r * LDW + 16 + half * 8];                \
      afr[i] = ua.v;                                                           \
    }                                                                          \
    _Pragma("unroll") for (int j = 0; j < 4; ++j) {                            \
      union { us8 h[2]; v16bf v; } ub;                                         \
      int n = wn + j * 16 + mr;                                                \
      ub.h[0] = *(const us8*)&Bs[BUF][n * LDW + half * 8];                     \
      ub.h[1] = *(const us8*)&Bs[BUF][n * LDW + 16 + half * 8];                \
      bfr[j] = ub.v;                                                           \
    }                                                                          \
    _Pragma("unroll") for (int i = 0; i < 2; ++i)                              \
      _Pragma("unroll") for (int j = 0; j < 4; ++j)                            \
        acc[i][j] = __builtin_amdgcn_wmma_f32_16x16x32_bf16(                   \
            false, afr[i], false, bfr[j], (short)0, acc[i][j], false, false);  \
  }

// C[M,N] = alpha*(A[M,K] @ B[K,N]) + bias ; optional exact-erf GELU epilogue.
// A: row-major (lda, contiguous cols). B: generic (row,col) strides, one of
// which must be 1. M % 128 == 0, K % 32 == 0; N bounds-checked (N%4 == 0).
__global__ __launch_bounds__(256)
void wmma_gemm_kernel(const float* __restrict__ A, const float* __restrict__ Bm,
                      const float* __restrict__ bias, float* __restrict__ C,
                      int Mrows, int Ncols, int Kd,
                      int lda, long long brs, long long bcs, int ldc,
                      float alpha, int gelu_flag)
{
  (void)Mrows;
  __shared__ __align__(16) unsigned short As[2][BM * LDW];
  __shared__ __align__(16) unsigned short Bs[2][BN * LDW];

  const int tid = threadIdx.x;
  const int mb  = blockIdx.y * BM;
  const int nb  = blockIdx.x * BN;

  const int lane = tid & 31;
  const int w    = tid >> 5;
  const int wm   = (w >> 1) * 32;   // 0,32,64,96
  const int wn   = (w & 1) * 64;    // 0,64
  const int mr   = lane & 15;
  const int half = lane >> 4;

  float4 areg[4], breg[4];

  v8f acc[2][4];
#pragma unroll
  for (int i = 0; i < 2; ++i)
#pragma unroll
    for (int j = 0; j < 4; ++j)
#pragma unroll
      for (int r = 0; r < 8; ++r) acc[i][j][r] = 0.0f;

  // prologue: stage k0 = 0
  GEMM_GLOAD(0);
  GEMM_LSTORE(0);
  __syncthreads();

  int buf = 0;
  for (int k0 = 0; k0 < Kd; k0 += BK) {
    const bool nxt = (k0 + BK) < Kd;
    if (nxt) GEMM_GLOAD(k0 + BK);     // global loads in flight during WMMAs
    GEMM_COMPUTE(buf);
    if (nxt) GEMM_LSTORE(buf ^ 1);
    __syncthreads();
    buf ^= 1;
  }

  // C layout: v8f elem r -> M = r + 8*half, N = lane&15.
#pragma unroll
  for (int j = 0; j < 4; ++j) {
    int col = nb + wn + j * 16 + mr;
    if (col >= Ncols) continue;
    float bv = bias ? bias[col] : 0.0f;
#pragma unroll
    for (int i = 0; i < 2; ++i) {
#pragma unroll
      for (int r = 0; r < 8; ++r) {
        int row = mb + wm + i * 16 + r + 8 * half;
        float vv = acc[i][j][r] * alpha + bv;
        if (gelu_flag) vv = 0.5f * vv * (1.0f + erff(vv * 0.70710678118654752f));
        C[(long long)row * ldc + col] = vv;
      }
    }
  }
}

// ---------------------------------------------------------------------------
// Tensor Data Mover demonstration (gfx1250 TDM: tensor_load_to_lds +
// s_wait_tensorcnt). Not launched: the production GEMM must convert fp32 ->
// bf16 while staging, which a raw DMA cannot do. D# fields per ISA 8.3/8.4.
// ---------------------------------------------------------------------------
__global__ void tdm_demo_kernel(const float* __restrict__ src, float* __restrict__ dst)
{
  __shared__ __align__(16) float tile[BK * LDW];
  unsigned long long ga = (unsigned long long)(const void*)src;

  u32x4 g0;
  g0[0] = 1u;                                   // count=1 (valid), user mode
  g0[1] = 0u;                                   // lds_addr = 0
  g0[2] = (unsigned)(ga & 0xffffffffu);         // global_addr[31:0]
  g0[3] = (unsigned)((ga >> 32) & 0x01ffffffu)  // global_addr[56:32]
          | (2u << 30);                         // type = 2 ("image")
  i32x8 g1;
  g1[0] = (2 << 16);                            // wg_mask=0, data_size=2 (4B)
  g1[1] = (int)((1024u & 0xffffu) << 16);       // tensor_dim0 low half
  g1[2] = (int)(1024u >> 16);                   // tensor_dim0 high half
  g1[3] = (32 << 16);                           // tile_dim0 = 32
  g1[4] = 0; g1[5] = 0; g1[6] = 0; g1[7] = 0;
  i32x4 g2; g2[0] = 0; g2[1] = 0; g2[2] = 0; g2[3] = 0;
  i32x4 g3; g3[0] = 0; g3[1] = 0; g3[2] = 0; g3[3] = 0;

#if defined(__clang_major__) && (__clang_major__ >= 23)
  i32x8 g4; g4[0] = 0; g4[1] = 0; g4[2] = 0; g4[3] = 0;
  g4[4] = 0; g4[5] = 0; g4[6] = 0; g4[7] = 0;
  __builtin_amdgcn_tensor_load_to_lds(g0, g1, g2, g3, g4, 0);
#else
  __builtin_amdgcn_tensor_load_to_lds(g0, g1, g2, g3, 0);
#endif
  __builtin_amdgcn_s_wait_tensorcnt(0);
  __syncthreads();
  dst[threadIdx.x] = tile[threadIdx.x];
}

// depthwise conv over sequence (K=7, pad 3), output already transposed to [B,S,D]
__global__ __launch_bounds__(256)
void dwconv_kernel(const float* __restrict__ x, const float* __restrict__ dw,
                   float* __restrict__ out, long long total)
{
  long long idx = (long long)blockIdx.x * blockDim.x + threadIdx.x;
  if (idx >= total) return;
  const int D_ = 1024, S_ = 2048;
  int d = (int)(idx % D_);
  int s = (int)((idx / D_) % S_);
  long long bbase = (idx / ((long long)D_ * S_)) * (long long)S_ * D_;
  float a = 0.0f;
#pragma unroll
  for (int t = 0; t < 7; ++t) {
    int ss = s + t - 3;
    if (ss >= 0 && ss < S_) a += x[bbase + (long long)ss * D_ + d] * dw[d * 7 + t];
  }
  out[idx] = a;
}

__global__ __launch_bounds__(256)
void mul_inplace_kernel(float* __restrict__ a, const float* __restrict__ b, long long total)
{
  long long idx = (long long)blockIdx.x * blockDim.x + threadIdx.x;
  if (idx < total) a[idx] *= b[idx];
}

// softmax over the 7 kernel taps, one thread per (b,s,h)
__global__ __launch_bounds__(256)
void softmax7_kernel(float* __restrict__ p, long long rows)
{
  long long r = (long long)blockIdx.x * blockDim.x + threadIdx.x;
  if (r >= rows) return;
  float* row = p + r * 7;
  float mx = row[0];
#pragma unroll
  for (int i = 1; i < 7; ++i) mx = fmaxf(mx, row[i]);
  float s = 0.0f, e[7];
#pragma unroll
  for (int i = 0; i < 7; ++i) { e[i] = expf(row[i] - mx); s += e[i]; }
  float inv = 1.0f / s;
#pragma unroll
  for (int i = 0; i < 7; ++i) row[i] = e[i] * inv;
}

// conv_out[b,s,h,dh] = sum_t co[b,s+t-3,h*64+dh] * kern[b,s,h,t] -> cat cols 512+
__global__ __launch_bounds__(256)
void dynconv_kernel(const float* __restrict__ co, const float* __restrict__ kern,
                    float* __restrict__ cat, long long total)
{
  long long idx = (long long)blockIdx.x * blockDim.x + threadIdx.x;
  if (idx >= total) return;
  const int DH_ = 64, H_ = 8, S_ = 2048, A_ = 512, D_ = 1024;
  int dh = (int)(idx % DH_);
  int h  = (int)((idx / DH_) % H_);
  int s  = (int)((idx / ((long long)DH_ * H_)) % S_);
  long long b = idx / ((long long)DH_ * H_ * S_);
  long long rowb = b * S_;
  const float* kr = kern + ((rowb + s) * H_ + h) * 7;
  float a = 0.0f;
#pragma unroll
  for (int t = 0; t < 7; ++t) {
    int ss = s + t - 3;
    if (ss >= 0 && ss < S_) a += co[(rowb + ss) * A_ + h * DH_ + dh] * kr[t];
  }
  cat[(rowb + s) * D_ + A_ + h * DH_ + dh] = a;
}

// masked row softmax, one block per row of 2048
__global__ __launch_bounds__(256)
void softmax_row_kernel(float* __restrict__ scores, const int* __restrict__ mask, int len)
{
  const int CNT = 8;  // len = 2048, 256 threads
  int row = blockIdx.x, tid = threadIdx.x;
  float* p = scores + (long long)row * len;
  float lv[CNT];
  float mx = -3.4028235e38f;
  for (int i = 0; i < CNT; ++i) {
    int j = i * 256 + tid;
    float v = p[j];
    if (!mask[j]) v += -3.4028235e38f;
    lv[i] = v;
    mx = fmaxf(mx, v);
  }
  __shared__ float red[256];
  red[tid] = mx; __syncthreads();
  for (int st = 128; st > 0; st >>= 1) {
    if (tid < st) red[tid] = fmaxf(red[tid], red[tid + st]);
    __syncthreads();
  }
  mx = red[0]; __syncthreads();
  float s = 0.0f;
  for (int i = 0; i < CNT; ++i) { lv[i] = expf(lv[i] - mx); s += lv[i]; }
  red[tid] = s; __syncthreads();
  for (int st = 128; st > 0; st >>= 1) {
    if (tid < st) red[tid] += red[tid + st];
    __syncthreads();
  }
  float inv = 1.0f / red[0];
  for (int i = 0; i < CNT; ++i) p[i * 256 + tid] = lv[i] * inv;
}

// out = LN(gin + resid) * g + b, row length 1024, one block per row
__global__ __launch_bounds__(256)
void ln_residual_kernel(const float* __restrict__ gin, const float* __restrict__ resid,
                        const float* __restrict__ g, const float* __restrict__ bta,
                        float* __restrict__ out)
{
  const int D_ = 1024, CNT = 4;
  int row = blockIdx.x, tid = threadIdx.x;
  long long base = (long long)row * D_;
  float lv[CNT], s = 0.0f;
  for (int i = 0; i < CNT; ++i) {
    int j = i * 256 + tid;
    float v = gin[base + j] + resid[base + j];
    lv[i] = v; s += v;
  }
  __shared__ float red[256];
  red[tid] = s; __syncthreads();
  for (int st = 128; st > 0; st >>= 1) {
    if (tid < st) red[tid] += red[tid + st];
    __syncthreads();
  }
  float mean = red[0] / (float)D_; __syncthreads();
  float s2 = 0.0f;
  for (int i = 0; i < CNT; ++i) { float d = lv[i] - mean; s2 += d * d; }
  red[tid] = s2; __syncthreads();
  for (int st = 128; st > 0; st >>= 1) {
    if (tid < st) red[tid] += red[tid + st];
    __syncthreads();
  }
  float inv = 1.0f / sqrtf(red[0] / (float)D_ + 1e-12f);
  for (int i = 0; i < CNT; ++i) {
    int j = i * 256 + tid;
    out[base + j] = (lv[i] - mean) * inv * g[j] + bta[j];
  }
}

static void launch_gemm(hipStream_t s, const float* A, const float* B, const float* bias,
                        float* C, int M, int N, int Kd, int lda,
                        long long brs, long long bcs, int ldc, float alpha, int gelu)
{
  dim3 grid((N + BN - 1) / BN, M / BM, 1);
  wmma_gemm_kernel<<<grid, dim3(256), 0, s>>>(A, B, bias, C, M, N, Kd, lda, brs, bcs,
                                              ldc, alpha, gelu);
}

extern "C" void kernel_launch(void* const* d_in, const int* in_sizes, int n_in,
                              void* d_out, int out_size, void* d_ws, size_t ws_size,
                              hipStream_t stream)
{
  (void)in_sizes; (void)n_in; (void)out_size; (void)ws_size;
  const int B = 2, S = 2048, D = 1024, H = 8, DH = 64, A = 512, F = 4096;
  const int M = B * S;  // 4096 token rows

  const float* x    = (const float*)d_in[0];
  const int*   am   = (const int*)  d_in[1];
  const float* wq   = (const float*)d_in[2];
  const float* bq   = (const float*)d_in[3];
  const float* wk   = (const float*)d_in[4];
  const float* bk   = (const float*)d_in[5];
  const float* wv   = (const float*)d_in[6];
  const float* bv   = (const float*)d_in[7];
  const float* dwt  = (const float*)d_in[8];
  const float* pw   = (const float*)d_in[9];
  const float* sepb = (const float*)d_in[10];
  const float* wck  = (const float*)d_in[11];
  const float* bck  = (const float*)d_in[12];
  const float* wco  = (const float*)d_in[13];
  const float* bco  = (const float*)d_in[14];
  const float* wao  = (const float*)d_in[15];
  const float* bao  = (const float*)d_in[16];
  const float* ln1g = (const float*)d_in[17];
  const float* ln1b = (const float*)d_in[18];
  const float* wi   = (const float*)d_in[19];
  const float* bi   = (const float*)d_in[20];
  const float* wo   = (const float*)d_in[21];
  const float* bo   = (const float*)d_in[22];
  const float* ln2g = (const float*)d_in[23];
  const float* ln2b = (const float*)d_in[24];

  // workspace bump allocation (floats)
  float* ws   = (float*)d_ws;
  float* q    = ws;                         // M*A
  float* kbuf = q    + (size_t)M * A;       // M*A
  float* vbuf = kbuf + (size_t)M * A;       // M*A
  float* dwo  = vbuf + (size_t)M * A;       // M*D (dead after sep -> reused as scores)
  float* sep  = dwo  + (size_t)M * D;       // M*A
  float* co   = sep  + (size_t)M * A;       // M*A
  float* kern = co   + (size_t)M * A;       // M*H*7 = M*56
  float* cat  = kern + (size_t)M * 56;      // M*D
  float* g1   = cat  + (size_t)M * D;       // M*D (gemm epilogue buffer, reused)
  float* h1   = g1   + (size_t)M * D;       // M*D
  float* ff   = h1   + (size_t)M * D;       // M*F
  float* scores = dwo;                      // S*S = M*D floats, alias

  // --- projections: q, k, v, co ---
  launch_gemm(stream, x, wq, bq, q,    M, A, D, D, A, 1, A, 1.0f, 0);
  launch_gemm(stream, x, wk, bk, kbuf, M, A, D, D, A, 1, A, 1.0f, 0);
  launch_gemm(stream, x, wv, bv, vbuf, M, A, D, D, A, 1, A, 1.0f, 0);
  launch_gemm(stream, x, wco, bco, co, M, A, D, D, A, 1, A, 1.0f, 0);

  // --- separable conv branch ---
  {
    long long total = (long long)M * D;
    dwconv_kernel<<<dim3((unsigned)((total + 255) / 256)), 256, 0, stream>>>(x, dwt, dwo, total);
  }
  launch_gemm(stream, dwo, pw, sepb, sep, M, A, D, D, A, 1, A, 1.0f, 0);
  {
    long long total = (long long)M * A;
    mul_inplace_kernel<<<dim3((unsigned)((total + 255) / 256)), 256, 0, stream>>>(sep, q, total);
  }
  launch_gemm(stream, sep, wck, bck, kern, M, 56, A, A, 56, 1, 56, 1.0f, 0);
  {
    long long rows = (long long)M * H;
    softmax7_kernel<<<dim3((unsigned)((rows + 255) / 256)), 256, 0, stream>>>(kern, rows);
  }
  {
    long long total = (long long)M * A;
    dynconv_kernel<<<dim3((unsigned)((total + 255) / 256)), 256, 0, stream>>>(co, kern, cat, total);
  }

  // --- self-attention, per (b,h): 16MB score matrix stays in L2 ---
  for (int b = 0; b < B; ++b) {
    for (int h = 0; h < H; ++h) {
      const float* qh = q    + (size_t)b * S * A + h * DH;
      const float* kh = kbuf + (size_t)b * S * A + h * DH;
      const float* vh = vbuf + (size_t)b * S * A + h * DH;
      float* ctx = cat + (size_t)b * S * D + h * DH;  // heads -> cat cols [0,512)
      // scores = (q @ k^T) / 8 : B addressed transposed via strides (brs=1)
      launch_gemm(stream, qh, kh, nullptr, scores, S, S, DH, A, 1LL, (long long)A, S,
                  0.125f, 0);
      softmax_row_kernel<<<dim3(S), 256, 0, stream>>>(scores, am + (size_t)b * S, S);
      // ctx = probs @ v
      launch_gemm(stream, scores, vh, nullptr, ctx, S, DH, S, S, (long long)A, 1LL, D,
                  1.0f, 0);
    }
  }

  // --- attention output + LN1 ---
  launch_gemm(stream, cat, wao, bao, g1, M, D, D, D, D, 1, D, 1.0f, 0);
  ln_residual_kernel<<<dim3(M), 256, 0, stream>>>(g1, x, ln1g, ln1b, h1);

  // --- FFN (GELU epilogue on first GEMM) + LN2 ---
  launch_gemm(stream, h1, wi, bi, ff, M, F, D, D, F, 1, F, 1.0f, 1);
  launch_gemm(stream, ff, wo, bo, g1, M, D, F, F, D, 1, D, 1.0f, 0);
  ln_residual_kernel<<<dim3(M), 256, 0, stream>>>(g1, h1, ln2g, ln2b, (float*)d_out);
}
